// Jastrow_84670985273665
// MI455X (gfx1250) — matrix operand order repaired and checked by reading, not compiled, not run on hardware
//
#include <hip/hip_runtime.h>
#include <math.h>

#define N_EL   1024
#define N_UP   512
#define WIDTH  64
#define D_EMB  256

// same-spin pairs: 2 * 512*511 = 523264 ; diff-spin: 2 * 512*512 = 524288
#define SAME_HALF   (512 * 511)
#define DIFF_HALF   (512 * 512)
#define N_SAME      (2 * SAME_HALF)
#define N_DIFF      (2 * DIFF_HALF)

typedef __attribute__((ext_vector_type(16))) __bf16 v16bf;
typedef __attribute__((ext_vector_type(8)))  float  v8f;

union BfVec {
    v16bf v;
    unsigned short u[16];
    unsigned int   w[8];
};

__device__ __forceinline__ unsigned short f32_to_bf16(float x) {
    unsigned int u = __float_as_uint(x);
    u += 0x7fffu + ((u >> 16) & 1u);       // round-to-nearest-even (init-time only)
    return (unsigned short)(u >> 16);
}

// pack two f32 -> one u32 holding 2 bf16 (round-to-nearest, ties away; cheap)
__device__ __forceinline__ unsigned int pack2_bf16(float lo, float hi) {
    unsigned int ul = __float_as_uint(lo) + 0x8000u;
    unsigned int uh = __float_as_uint(hi) + 0x8000u;
    return (ul >> 16) | (uh & 0xffff0000u);
}

// ---- branchless, hardware-native transcendentals -------------------------
__device__ __forceinline__ float fast_rcp(float x)  { return __builtin_amdgcn_rcpf(x); }
__device__ __forceinline__ float fast_sqrt(float x) { return __builtin_amdgcn_sqrtf(x); }

__device__ __forceinline__ float fast_tanh(float x) {
#if __has_builtin(__builtin_amdgcn_tanhf)
    return __builtin_amdgcn_tanhf(x);          // native V_TANH_F32 on gfx1250
#else
    const float e = __expf(2.0f * x);
    return (e - 1.0f) * __builtin_amdgcn_rcpf(e + 1.0f);
#endif
}

// ---------------------------------------------------------------------------
// Pair kernel: 16 pairs per wave-iteration.
//   layer0 (4->64): f32 VALU, output packed directly in WMMA bf16 A-layout
//   layer1 (64->64): 8x v_wmma_f32_16x16x32_bf16 per 16 pairs
//   layer2 (64->1) + tanh: pipelined tanh + tree reduce
//   cusp term: f32 VALU on lanes 0..15
// ---------------------------------------------------------------------------
__global__ void __launch_bounds__(256, 1)
pair_mlp_kernel(const float* __restrict__ el,
                const float* __restrict__ Aparam,
                const float* __restrict__ w0,   // [4][64]
                const float* __restrict__ b0,   // [64]
                const float* __restrict__ w1,   // [64][64]
                const float* __restrict__ b1,   // [64]
                const float* __restrict__ w2,   // [64]
                float* __restrict__ ws,
                int mode, int nIter, int cuspSlot, int mlpSlot)
{
    __shared__ float s_w0[WIDTH * 4];   // transposed: [k][j], float4 per column
    __shared__ float s_b0[WIDTH];

    const int tid  = blockIdx.x * blockDim.x + threadIdx.x;
    const int lane = threadIdx.x & 31;
    const int wave = tid >> 5;
    const int nWaves = (gridDim.x * blockDim.x) >> 5;

    // stage W0 transposed + b0 into LDS (once per block)
    if (threadIdx.x < 256) {
        int k = threadIdx.x >> 2, j = threadIdx.x & 3;
        s_w0[threadIdx.x] = w0[j * WIDTH + k];
    }
    if (threadIdx.x < WIDTH) s_b0[threadIdx.x] = b0[threadIdx.x];
    __syncthreads();

    // cusp width F = sqrt(2 * softplus(A))
    const float Aval = Aparam[0];
    const float Finv = fast_rcp(fast_sqrt(2.0f * __logf(1.0f + __expf(Aval))));

    const int n16   = lane & 15;            // output column within a 16-tile
    const int klo8  = (lane < 16) ? 0 : 8;  // A-layout K sub-block for this half-wave
    const int klo16 = (lane < 16) ? 0 : 16; // unused (B preload also uses klo8)

    // --- preload W1 in bf16 B-operand layout (8 x v16bf), b1 / W2 slices ---
    BfVec Bop[2][4];
    float b1n[4], w2n[4];
#pragma unroll
    for (int c = 0; c < 2; ++c)
#pragma unroll
        for (int t = 0; t < 4; ++t)
#pragma unroll
            for (int e = 0; e < 16; ++e)
                Bop[c][t].u[e] = f32_to_bf16(w1[(32 * c + klo8 + (e & 7) + ((e >> 3) << 4)) * WIDTH + 16 * t + n16]);
#pragma unroll
    for (int t = 0; t < 4; ++t) {
        b1n[t] = b1[16 * t + n16];
        w2n[t] = w2[16 * t + n16];
    }

    float accM = 0.0f;   // raw sum of MLP outputs
    float accC = 0.0f;   // raw sum of expm1(-d/F)/d

    for (int it = wave; it < nIter; it += nWaves) {
        const int p = it * 16 + n16;        // pair index for this lane (dup in both halves)

        // ---- decode (i, j) for this pair ----
        int i, j;
        if (mode == 0) {
            int q = p, off = 0;
            if (q >= SAME_HALF) { q -= SAME_HALF; off = N_UP; }
            int ii = q / 511, jj = q % 511;
            jj += (jj >= ii);
            i = ii + off; j = jj + off;
        } else {
            if (p < DIFF_HALF) { i = p >> 9;            j = N_UP + (p & 511); }
            else { int q = p - DIFF_HALF; i = N_UP + (q >> 9); j = q & 511; }
        }

        // ---- features ----
        const float ex = el[3 * i + 0] - el[3 * j + 0];
        const float ey = el[3 * i + 1] - el[3 * j + 1];
        const float ez = el[3 * i + 2] - el[3 * j + 2];
        const float r    = fast_sqrt(ex * ex + ey * ey + ez * ez);
        const float rinv = fast_rcp(r);
        const float lr   = __logf(1.0f + r);
        const float sc   = lr * rinv;
        const float f0 = ex * sc, f1 = ey * sc, f2 = ez * sc, f3 = lr;

        if (lane < 16) accC += (__expf(-r * Finv) - 1.0f) * rinv;

        // ---- layer 0: each lane computes the 32 h0 columns its half-wave
        //      owns, packed pairwise straight into bf16 A-operand words ----
        BfVec A[2];
#pragma unroll
        for (int g = 0; g < 4; ++g) {
            const int kbase = g * 16 + klo8;
#pragma unroll
            for (int e2 = 0; e2 < 4; ++e2) {
                const int k0 = kbase + 2 * e2;
                const float4 wv0 = ((const float4*)s_w0)[k0];
                const float4 wv1 = ((const float4*)s_w0)[k0 + 1];
                float p0 = s_b0[k0]     + f0 * wv0.x + f1 * wv0.y + f2 * wv0.z + f3 * wv0.w;
                float p1 = s_b0[k0 + 1] + f0 * wv1.x + f1 * wv1.y + f2 * wv1.z + f3 * wv1.w;
                A[g >> 1].w[(g & 1) * 4 + e2] = pack2_bf16(fast_tanh(p0), fast_tanh(p1));
            }
        }

        // ---- layer 1 (WMMA bf16, f32 accum) + layer 2 fused reduce ----
#pragma unroll
        for (int t = 0; t < 4; ++t) {
            const float bb = b1n[t];
            v8f cc = {bb, bb, bb, bb, bb, bb, bb, bb};
            cc = __builtin_amdgcn_wmma_f32_16x16x32_bf16(false, A[0].v, false, Bop[0][t].v,
                                                         (short)0, cc, false, false);
            cc = __builtin_amdgcn_wmma_f32_16x16x32_bf16(false, A[1].v, false, Bop[1][t].v,
                                                         (short)0, cc, false, false);
            // independent tanhs (pipeline through trans unit), then tree-sum
            float t0 = fast_tanh(cc[0]), t1 = fast_tanh(cc[1]);
            float t2 = fast_tanh(cc[2]), t3 = fast_tanh(cc[3]);
            float t4 = fast_tanh(cc[4]), t5 = fast_tanh(cc[5]);
            float t6 = fast_tanh(cc[6]), t7 = fast_tanh(cc[7]);
            const float s0 = (t0 + t1) + (t2 + t3);
            const float s1 = (t4 + t5) + (t6 + t7);
            accM += w2n[t] * (s0 + s1);
        }
    }

    // ---- wave reduction + one atomic per wave ----
#pragma unroll
    for (int off = 16; off > 0; off >>= 1) {
        accM += __shfl_down(accM, off, 32);
        accC += __shfl_down(accC, off, 32);
    }
    if (lane == 0) {
        atomicAdd(&ws[mlpSlot],  accM);
        atomicAdd(&ws[cuspSlot], accC);
    }
}

// ---------------------------------------------------------------------------
// Embedding MLP: 1024 rows x (256->64->64->2); < 1% of total FLOPs.
// ---------------------------------------------------------------------------
__global__ void emb_mlp_kernel(const float* __restrict__ emb,
                               const float* __restrict__ We0,
                               const float* __restrict__ be0,
                               const float* __restrict__ We1,
                               const float* __restrict__ be1,
                               const float* __restrict__ We2,
                               float* __restrict__ ws)
{
    const int row = blockIdx.x * blockDim.x + threadIdx.x;
    if (row >= N_EL) return;

    float h0[WIDTH];
#pragma unroll
    for (int k = 0; k < WIDTH; ++k) h0[k] = be0[k];

    const float* er = emb + row * D_EMB;
    for (int jj = 0; jj < D_EMB; ++jj) {
        const float v = er[jj];
        const float* wr = We0 + jj * WIDTH;
#pragma unroll
        for (int k = 0; k < WIDTH; ++k) h0[k] += v * wr[k];
    }
#pragma unroll
    for (int k = 0; k < WIDTH; ++k) h0[k] = fast_tanh(h0[k]);

    float o0 = 0.0f, o1 = 0.0f;
    for (int k2 = 0; k2 < WIDTH; ++k2) {
        float s = be1[k2];
#pragma unroll
        for (int k = 0; k < WIDTH; ++k) s += h0[k] * We1[k * WIDTH + k2];
        const float t = fast_tanh(s);
        o0 += t * We2[2 * k2 + 0];
        o1 += t * We2[2 * k2 + 1];
    }
    atomicAdd(&ws[4], o0);
    atomicAdd(&ws[5], o1);
}

// ---------------------------------------------------------------------------
// Finalize: single thread assembles (sign, logpsi).
// ws: [0]=cusp_same [1]=mlp_same [2]=cusp_diff [3]=mlp_diff [4]=e0 [5]=e1
// ---------------------------------------------------------------------------
__global__ void finalize_kernel(const float* __restrict__ ws,
                                const float* __restrict__ A_same,
                                const float* __restrict__ A_diff,
                                const float* __restrict__ scale_same,
                                const float* __restrict__ scale_diff,
                                const float* __restrict__ be2,
                                const float* __restrict__ mlp_scale,
                                const float* __restrict__ log_bias,
                                float* __restrict__ out)
{
    const float A_s = log1pf(expf(A_same[0]));
    const float A_d = log1pf(expf(A_diff[0]));

    const float j0 = mlp_scale[0] * (ws[4] + (float)N_EL * be2[0]);
    const float j1 = mlp_scale[1] * (ws[5] + (float)N_EL * be2[1]) + (float)N_EL * log_bias[0];

    float logpsi = A_s * ws[0] + A_d * ws[2]
                 + scale_same[0] * ws[1] + scale_diff[0] * ws[3]
                 + j0 + logf(fabsf(j1));

    out[0] = (j1 > 0.0f) ? 1.0f : ((j1 < 0.0f) ? -1.0f : 0.0f);  // sign
    out[1] = logpsi;
}

extern "C" void kernel_launch(void* const* d_in, const int* in_sizes, int n_in,
                              void* d_out, int out_size, void* d_ws, size_t ws_size,
                              hipStream_t stream)
{
    const float* electrons  = (const float*)d_in[0];
    const float* embeddings = (const float*)d_in[1];
    const float* A_same     = (const float*)d_in[2];
    const float* A_diff     = (const float*)d_in[3];
    const float* Ws0_same   = (const float*)d_in[4];
    const float* bs0_same   = (const float*)d_in[5];
    const float* Ws1_same   = (const float*)d_in[6];
    const float* bs1_same   = (const float*)d_in[7];
    const float* Ws2_same   = (const float*)d_in[8];
    const float* Ws0_diff   = (const float*)d_in[9];
    const float* bs0_diff   = (const float*)d_in[10];
    const float* Ws1_diff   = (const float*)d_in[11];
    const float* bs1_diff   = (const float*)d_in[12];
    const float* Ws2_diff   = (const float*)d_in[13];
    const float* scale_same = (const float*)d_in[14];
    const float* scale_diff = (const float*)d_in[15];
    const float* We0        = (const float*)d_in[16];
    const float* be0        = (const float*)d_in[17];
    const float* We1        = (const float*)d_in[18];
    const float* be1        = (const float*)d_in[19];
    const float* We2        = (const float*)d_in[20];
    const float* be2        = (const float*)d_in[21];
    const float* mlp_scale  = (const float*)d_in[22];
    const float* log_bias   = (const float*)d_in[23];

    float* ws  = (float*)d_ws;
    float* out = (float*)d_out;

    hipMemsetAsync(d_ws, 0, 6 * sizeof(float), stream);

    // embedding MLP: 1024 rows, one thread each
    emb_mlp_kernel<<<4, 256, 0, stream>>>(embeddings, We0, be0, We1, be1, We2, ws);

    // pair MLPs: 16 pairs per wave-iteration, grid-stride
    const int blocks = 1024, threads = 256;
    pair_mlp_kernel<<<blocks, threads, 0, stream>>>(
        electrons, A_same, Ws0_same, bs0_same, Ws1_same, bs1_same, Ws2_same,
        ws, /*mode=*/0, N_SAME / 16, /*cuspSlot=*/0, /*mlpSlot=*/1);
    pair_mlp_kernel<<<blocks, threads, 0, stream>>>(
        electrons, A_diff, Ws0_diff, bs0_diff, Ws1_diff, bs1_diff, Ws2_diff,
        ws, /*mode=*/1, N_DIFF / 16, /*cuspSlot=*/2, /*mlpSlot=*/3);

    finalize_kernel<<<1, 1, 0, stream>>>(ws, A_same, A_diff, scale_same, scale_diff,
                                         be2, mlp_scale, log_bias, out);
}
